// GraphTransformerModel_89180700934259
// MI455X (gfx1250) — compile-verified, matrix operand order checked by reading, not run on hardware
//
#include <hip/hip_runtime.h>

// ---------------------------------------------------------------------------
// Types / helpers for CDNA5 WMMA (gfx1250, wave32)
// ---------------------------------------------------------------------------
typedef __attribute__((ext_vector_type(16))) __bf16 v16bf;
typedef __attribute__((ext_vector_type(8)))  __bf16 v8bf;
typedef __attribute__((ext_vector_type(8)))  float  v8f;

__device__ __forceinline__ unsigned short f2bf_bits(float f) {
    unsigned u = __float_as_uint(f);
    unsigned rnd = 0x7FFFu + ((u >> 16) & 1u);   // round-to-nearest-even
    return (unsigned short)((u + rnd) >> 16);
}

__device__ __forceinline__ v8f wmma_bf16(v16bf a, v16bf b, v8f c) {
    // v_wmma_f32_16x16x32_bf16  (neg_a, A, neg_b, B, c_mod, C, reuse_a, reuse_b)
    return __builtin_amdgcn_wmma_f32_16x16x32_bf16(false, a, false, b, (short)0, c, false, false);
}

__device__ __forceinline__ v16bf cat16(v8bf lo, v8bf hi) {
    return __builtin_shufflevector(lo, hi, 0, 1, 2, 3, 4, 5, 6, 7, 8, 9, 10, 11, 12, 13, 14, 15);
}

// A fragment (16x32 bf16, row-major, leading dim lda): per the ISA layout a
// lane's 16 elements are two runs of 8 consecutive K values -> 2x 16B loads.
__device__ __forceinline__ v16bf load_a_frag(const unsigned short* A, int lda, int row, int kk) {
    int lane = threadIdx.x & 31;
    int r    = row + (lane & 15);
    int half = (lane >> 4) * 8;
    const unsigned short* base = A + (size_t)r * lda + kk + half;
    v8bf lo = *(const v8bf*)(base);
    v8bf hi = *(const v8bf*)(base + 16);
    return cat16(lo, hi);
}

// B fragment from fragment-major swizzled panel: lane's 16 values contiguous.
// Panel layout: [ktile][lane][16] bf16 (512 bf16 per ktile).
__device__ __forceinline__ v16bf load_b_frag_sw(const unsigned short* P, int kt) {
    int lane = threadIdx.x & 31;
    const unsigned short* base = P + ((kt * 32 + lane) << 4);
    v8bf lo = *(const v8bf*)(base);
    v8bf hi = *(const v8bf*)(base + 8);
    return cat16(lo, hi);
}

#define NN    8192
#define HH    128
#define KNB   32
#define NB    16
#define GSZ   512
#define FFD   512

// ---------------------------------------------------------------------------
// f32 -> bf16 (row-major), used for activations feeding the A side
// ---------------------------------------------------------------------------
__global__ void k_f2bf(const float* __restrict__ src, unsigned short* __restrict__ dst, int count) {
    int i = blockIdx.x * blockDim.x + threadIdx.x;
    if (i < count) dst[i] = f2bf_bits(src[i]);
}

// ---------------------------------------------------------------------------
// Weight swizzle: f32 B[Kd][Nd] -> bf16 fragment-major [nt][kt][lane][e]
// (dst idx = ((nt*(Kd/32) + kt)*32 + lane)*16 + e)
// ---------------------------------------------------------------------------
__global__ void k_swizzleB(const float* __restrict__ src, unsigned short* __restrict__ dst,
                           int Kd, int Nd) {
    int idx = blockIdx.x * blockDim.x + threadIdx.x;
    if (idx >= Kd * Nd) return;
    int KT   = Kd >> 5;
    int e    = idx & 15;
    int lane = (idx >> 4) & 31;
    int kt   = (idx >> 9) % KT;
    int nt   = (idx >> 9) / KT;
    int j    = lane & 15, half = lane >> 4;
    int grp  = e >> 3;
    int kl   = grp * 16 + half * 8 + (e & 7);
    int k    = kt * 32 + kl;
    int n    = nt * 16 + j;
    dst[idx] = f2bf_bits(src[k * Nd + n]);
}

// ---------------------------------------------------------------------------
// node embedding gather: h = node_embed[x]  (f32 + bf16 copies)
// ---------------------------------------------------------------------------
__global__ void k_embed(const int* __restrict__ x, const float* __restrict__ emb,
                        float* __restrict__ h, unsigned short* __restrict__ hb) {
    int idx = blockIdx.x * blockDim.x + threadIdx.x;
    if (idx >= NN * HH) return;
    int n = idx >> 7, c = idx & 127;
    float v = emb[x[n] * HH + c];
    h[idx]  = v;
    hb[idx] = f2bf_bits(v);
}

// ---------------------------------------------------------------------------
// KNN within each 512-node graph: K=32 nearest with d2 <= 64, excluding self.
// ---------------------------------------------------------------------------
__global__ void k_knn(const float* __restrict__ pos, int* __restrict__ nbr,
                      float* __restrict__ elen, float* __restrict__ emask) {
    __shared__ float d2s[GSZ];
    __shared__ float rv[256];
    __shared__ int   ri[256];
    int n = blockIdx.x, t = threadIdx.x;
    int base = (n >> 9) << 9;                         // graph start
    float px = pos[n * 3], py = pos[n * 3 + 1], pz = pos[n * 3 + 2];
    for (int j = t; j < GSZ; j += 256) {
        int c = base + j;
        float dx = px - pos[c * 3], dy = py - pos[c * 3 + 1], dz = pz - pos[c * 3 + 2];
        float d2 = dx * dx + dy * dy + dz * dz;
        bool ok = (c != n) && (d2 <= 64.0f);
        d2s[j] = ok ? d2 : 1e30f;
    }
    __syncthreads();
    for (int kSel = 0; kSel < KNB; ++kSel) {
        float bv = 1e30f; int bi = base;
        for (int j = t; j < GSZ; j += 256) {
            float v = d2s[j];
            if (v < bv) { bv = v; bi = base + j; }
        }
        rv[t] = bv; ri[t] = bi;
        __syncthreads();
        for (int s = 128; s > 0; s >>= 1) {
            if (t < s && rv[t + s] < rv[t]) { rv[t] = rv[t + s]; ri[t] = ri[t + s]; }
            __syncthreads();
        }
        if (t == 0) {
            float v = rv[0]; int idx = ri[0];
            if (v < 1e29f) {
                nbr[n * KNB + kSel]   = idx;
                elen[n * KNB + kSel]  = sqrtf(fmaxf(v, 1e-12f));
                emask[n * KNB + kSel] = 1.0f;
                d2s[idx - base] = 1e30f;
            } else {
                nbr[n * KNB + kSel] = n; elen[n * KNB + kSel] = 1.0f; emask[n * KNB + kSel] = 0.0f;
            }
        }
        __syncthreads();
    }
}

// ---------------------------------------------------------------------------
// bf16 WMMA GEMM: C[M,Nd] = A[M,Kd] * Bsw   (Bsw = fragment-swizzled weights)
// block = 128 thr (4 waves); block tile = 128 rows x 16 cols; each wave does
// TWO 16x16 M-tiles with independent accumulators sharing one B fragment.
// MODE 0: C = A*B (f32)   MODE 1: C = A*B + Res (f32)   MODE 2: Cb = relu(A*B) (bf16)
// ---------------------------------------------------------------------------
template <int MODE>
__global__ void k_gemm(const unsigned short* __restrict__ A, const unsigned short* __restrict__ Bsw,
                       const float* __restrict__ Res, float* __restrict__ Cf,
                       unsigned short* __restrict__ Cb, int M, int Nd, int Kd) {
    __shared__ alignas(16) unsigned short Bs[512 * 16];   // swizzled K-panel (<=16 KB)
    int nt = blockIdx.y;
    // stage swizzled panel: contiguous uint4 copy of Kd*16 bf16
    {
        const uint4* src = (const uint4*)(Bsw + (size_t)nt * Kd * 16);
        uint4* dst = (uint4*)Bs;
        int n16 = Kd * 2;                                 // Kd*16 shorts / 8 per uint4
        for (int i = threadIdx.x; i < n16; i += blockDim.x) dst[i] = src[i];
    }
    __syncthreads();

    int wave = threadIdx.x >> 5;
    int row0 = blockIdx.x * 128 + wave * 16;
    int row1 = row0 + 64;
    v8f acc0 = {}, acc1 = {};
    for (int kk = 0; kk < Kd; kk += 32) {
        if (kk + 32 < Kd) {
            __builtin_prefetch(A + (size_t)row0 * Kd + kk + 32, 0, 0);
            __builtin_prefetch(A + (size_t)row1 * Kd + kk + 32, 0, 0);
        }
        v16bf b  = load_b_frag_sw(Bs, kk >> 5);
        v16bf a0 = load_a_frag(A, Kd, row0, kk);
        v16bf a1 = load_a_frag(A, Kd, row1, kk);
        acc0 = wmma_bf16(a0, b, acc0);
        acc1 = wmma_bf16(a1, b, acc1);
    }
    int lane = threadIdx.x & 31;
    int nc  = nt * 16 + (lane & 15);
    int mb0 = row0 + (lane >> 4) * 8;
    int mb1 = row1 + (lane >> 4) * 8;
#pragma unroll
    for (int i = 0; i < 8; ++i) {
        float v0 = acc0[i], v1 = acc1[i];
        size_t o0 = (size_t)(mb0 + i) * Nd + nc;
        size_t o1 = (size_t)(mb1 + i) * Nd + nc;
        if (MODE == 0)      { Cf[o0] = v0;            Cf[o1] = v1; }
        else if (MODE == 1) { Cf[o0] = v0 + Res[o0];  Cf[o1] = v1 + Res[o1]; }
        else                { Cb[o0] = f2bf_bits(v0 > 0.0f ? v0 : 0.0f);
                              Cb[o1] = f2bf_bits(v1 > 0.0f ? v1 : 0.0f); }
    }
}

// ---------------------------------------------------------------------------
// Fused per-node attention: RBF(elen) -> LDS bf16, WMMA x We(swizzled) -> e
// (LDS f32), edge-modulated scores, masked softmax over K, message aggregation.
// block = 128 thr (4 waves) per node.  e never touches HBM (saves 134 MB).
// Each wave: hoists 8 A fragments (4 kt x 2 mt) into registers once, then
// runs its two n-tiles with two interleaved accumulators (no WMMA RAW stalls).
// ---------------------------------------------------------------------------
__global__ void k_attn(const float* __restrict__ qp, const float* __restrict__ kp,
                       const float* __restrict__ vp, const unsigned short* __restrict__ WeSw,
                       const int* __restrict__ nbr, const float* __restrict__ elen,
                       const float* __restrict__ emask, float* __restrict__ msg) {
    __shared__ alignas(16) unsigned short sWe[HH * HH];  // 32 KB, fragment-major
    __shared__ alignas(16) unsigned short sA[KNB * HH];  //  8 KB (eattr bf16, row-major)
    __shared__ float          sE[KNB * HH];              // 16 KB (e f32)
    __shared__ float          sSc[4 * KNB];
    __shared__ float          sSum[4];
    __shared__ int            sNb[KNB];
    __shared__ float          sMask[KNB];
    int n = blockIdx.x, t = threadIdx.x;

    {   // stage swizzled We (contiguous 128-bit copies)
        const uint4* src = (const uint4*)WeSw;
        uint4* dst = (uint4*)sWe;
        for (int i = t; i < HH * HH / 8; i += 128) dst[i] = src[i];
    }
    if (t < KNB) { sNb[t] = nbr[n * KNB + t]; sMask[t] = emask[n * KNB + t]; }
    // eattr[k][h] = exp(-(elen - mu_h)^2), mu = linspace(3, 8, 128)
    for (int idx = t; idx < KNB * HH; idx += 128) {
        int k = idx >> 7, h = idx & 127;
        float el = elen[n * KNB + k];
        float mu = 3.0f + 5.0f * (float)h * (1.0f / 127.0f);
        float d  = el - mu;
        sA[idx] = f2bf_bits(__expf(-d * d));
    }
    __syncthreads();

    int wave = t >> 5;
    {
        // hoist all A fragments for this wave: [kt][mt]
        v16bf af[4][2];
#pragma unroll
        for (int kt = 0; kt < 4; ++kt) {
            af[kt][0] = load_a_frag(sA, HH, 0,  kt * 32);
            af[kt][1] = load_a_frag(sA, HH, 16, kt * 32);
        }
        int lane = t & 31;
#pragma unroll
        for (int ni = 0; ni < 2; ++ni) {                  // n-tiles: wave, wave+4
            int nt = wave + ni * 4;
            const unsigned short* panel = sWe + nt * (4 * 32 * 16);   // [kt][lane][16]
            v8f acc0 = {}, acc1 = {};
#pragma unroll
            for (int kt = 0; kt < 4; ++kt) {
                v16bf b = load_b_frag_sw(panel, kt);
                acc0 = wmma_bf16(af[kt][0], b, acc0);     // mt = 0
                acc1 = wmma_bf16(af[kt][1], b, acc1);     // mt = 1
            }
            int nc = nt * 16 + (lane & 15);
            int mb = (lane >> 4) * 8;
#pragma unroll
            for (int i = 0; i < 8; ++i) {
                sE[(mb + i) * HH + nc]        = acc0[i];
                sE[(16 + mb + i) * HH + nc]   = acc1[i];
            }
        }
    }
    __syncthreads();

    // scores: t -> (k, nh)
    {
        int k = t >> 2, nh = t & 3;
        int nbk = sNb[k];
        float s = 0.0f;
#pragma unroll 8
        for (int dh = 0; dh < 32; ++dh) {
            int h = nh * 32 + dh;
            s += qp[n * HH + h] * kp[nbk * HH + h] * sE[k * HH + h];
        }
        const float scale = 0.17677669529663689f;     // 1/sqrt(32)
        sSc[nh * KNB + k] = (sMask[k] > 0.5f) ? s * scale : -1e9f;
    }
    __syncthreads();
    if (t < 4) {
        float mx = -1e30f;
        for (int j = 0; j < KNB; ++j) mx = fmaxf(mx, sSc[t * KNB + j]);
        float sm = 0.0f;
        for (int j = 0; j < KNB; ++j) { float e = __expf(sSc[t * KNB + j] - mx); sSc[t * KNB + j] = e; sm += e; }
        sSum[t] = sm;
    }
    __syncthreads();
    // message: t = h
    {
        int h = t, nh = h >> 5;
        float inv = 1.0f / sSum[nh];
        float m = 0.0f;
#pragma unroll 8
        for (int k = 0; k < KNB; ++k) m += sSc[nh * KNB + k] * vp[sNb[k] * HH + h];
        msg[n * HH + h] = m * inv;
    }
}

// ---------------------------------------------------------------------------
// BatchNorm over axis 0 (N=8192), per column
// ---------------------------------------------------------------------------
__global__ void k_bnstats(const float* __restrict__ X, float* __restrict__ mean,
                          float* __restrict__ invstd) {
    __shared__ float s1[256], s2[256];
    int c = blockIdx.x, t = threadIdx.x;
    float a = 0.0f, b = 0.0f;
    for (int r = t; r < NN; r += 256) { float v = X[r * HH + c]; a += v; b += v * v; }
    s1[t] = a; s2[t] = b;
    __syncthreads();
    for (int s = 128; s > 0; s >>= 1) {
        if (t < s) { s1[t] += s1[t + s]; s2[t] += s2[t + s]; }
        __syncthreads();
    }
    if (t == 0) {
        float m = s1[0] * (1.0f / NN);
        float v = s2[0] * (1.0f / NN) - m * m;
        mean[c] = m; invstd[c] = rsqrtf(v + 1e-5f);
    }
}

__global__ void k_bnapply(const float* __restrict__ X, const float* __restrict__ mean,
                          const float* __restrict__ invstd, const float* __restrict__ g,
                          const float* __restrict__ b, float* __restrict__ Y,
                          unsigned short* __restrict__ Yb) {
    int idx = blockIdx.x * blockDim.x + threadIdx.x;
    if (idx >= NN * HH) return;
    int c = idx & 127;
    float y = (X[idx] - mean[c]) * invstd[c] * g[c] + b[c];
    Y[idx] = y;
    if (Yb) Yb[idx] = f2bf_bits(y);
}

// ---------------------------------------------------------------------------
// Fused segment-sum pooling + readout head: one block per graph
// ---------------------------------------------------------------------------
__global__ void k_pool_readout(const float* __restrict__ h2, const float* __restrict__ Wr1,
                               const float* __restrict__ br1, const float* __restrict__ Wr2,
                               const float* __restrict__ br2, float* __restrict__ out) {
    __shared__ float pooled[HH];
    __shared__ float red[HH];
    int b = blockIdx.x, t = threadIdx.x;     // 128 threads
    int base = b * GSZ;
    float s = 0.0f;
    for (int r = 0; r < GSZ; ++r) s += h2[(base + r) * HH + t];
    pooled[t] = s;
    __syncthreads();
    float a = br1[t];
    for (int h = 0; h < HH; ++h) a += pooled[h] * Wr1[h * HH + t];
    a = fmaxf(a, 0.0f);
    red[t] = a * Wr2[t];
    __syncthreads();
    for (int s2 = 64; s2 > 0; s2 >>= 1) {
        if (t < s2) red[t] += red[t + s2];
        __syncthreads();
    }
    if (t == 0) out[b] = red[0] + br2[0];
}

// ---------------------------------------------------------------------------
// Host-side orchestration
// ---------------------------------------------------------------------------
extern "C" void kernel_launch(void* const* d_in, const int* in_sizes, int n_in,
                              void* d_out, int out_size, void* d_ws, size_t ws_size,
                              hipStream_t stream) {
    (void)in_sizes; (void)n_in; (void)out_size;
    const float* pos   = (const float*)d_in[0];
    const int*   x     = (const int*)d_in[1];
    // d_in[2] = batch (unused: graphs are contiguous 512-node blocks)
    const float* emb   = (const float*)d_in[3];
    const float* Wq    = (const float*)d_in[4];
    const float* Wk    = (const float*)d_in[5];
    const float* Wv    = (const float*)d_in[6];
    const float* We    = (const float*)d_in[7];
    const float* Wo    = (const float*)d_in[8];
    const float* bn1g  = (const float*)d_in[9];
    const float* bn1b  = (const float*)d_in[10];
    const float* Wff1  = (const float*)d_in[11];
    const float* Wff2  = (const float*)d_in[12];
    const float* bn2g  = (const float*)d_in[13];
    const float* bn2b  = (const float*)d_in[14];
    const float* Wr1   = (const float*)d_in[15];
    const float* br1   = (const float*)d_in[16];
    const float* Wr2   = (const float*)d_in[17];
    const float* br2   = (const float*)d_in[18];
    float* out = (float*)d_out;

    // --- carve workspace ---
    char* p = (char*)d_ws;
    auto alloc = [&](size_t bytes) -> void* {
        void* r = (void*)p;
        p += (bytes + 255) & ~(size_t)255;
        return r;
    };
    float*          h     = (float*)alloc((size_t)NN * HH * 4);
    unsigned short* hb    = (unsigned short*)alloc((size_t)NN * HH * 2);
    unsigned short* WqS   = (unsigned short*)alloc(HH * HH * 2);
    unsigned short* WkS   = (unsigned short*)alloc(HH * HH * 2);
    unsigned short* WvS   = (unsigned short*)alloc(HH * HH * 2);
    unsigned short* WeS   = (unsigned short*)alloc(HH * HH * 2);
    unsigned short* WoS   = (unsigned short*)alloc(HH * HH * 2);
    unsigned short* Wff1S = (unsigned short*)alloc(HH * FFD * 2);
    unsigned short* Wff2S = (unsigned short*)alloc(FFD * HH * 2);
    float*          q     = (float*)alloc((size_t)NN * HH * 4);
    float*          kprj  = (float*)alloc((size_t)NN * HH * 4);
    float*          vprj  = (float*)alloc((size_t)NN * HH * 4);
    int*            nbr   = (int*)alloc((size_t)NN * KNB * 4);
    float*          elen  = (float*)alloc((size_t)NN * KNB * 4);
    float*          emask = (float*)alloc((size_t)NN * KNB * 4);
    float*          msg   = (float*)alloc((size_t)NN * HH * 4);
    unsigned short* msgb  = (unsigned short*)alloc((size_t)NN * HH * 2);
    float*          t1    = (float*)alloc((size_t)NN * HH * 4);
    float*          h1    = (float*)alloc((size_t)NN * HH * 4);
    unsigned short* h1b   = (unsigned short*)alloc((size_t)NN * HH * 2);
    unsigned short* ffb   = (unsigned short*)alloc((size_t)NN * FFD * 2);
    float*          t2    = (float*)alloc((size_t)NN * HH * 4);
    float*          h2    = (float*)alloc((size_t)NN * HH * 4);
    float*          mean1 = (float*)alloc(HH * 4);
    float*          istd1 = (float*)alloc(HH * 4);
    float*          mean2 = (float*)alloc(HH * 4);
    float*          istd2 = (float*)alloc(HH * 4);
    if ((size_t)(p - (char*)d_ws) > ws_size) return;   // workspace too small

    const int EW = NN * HH;
    dim3 ewg((EW + 255) / 256), ewb(256);

    // 1) weights -> fragment-swizzled bf16
    k_swizzleB<<<(HH * HH + 255) / 256, 256, 0, stream>>>(Wq, WqS, HH, HH);
    k_swizzleB<<<(HH * HH + 255) / 256, 256, 0, stream>>>(Wk, WkS, HH, HH);
    k_swizzleB<<<(HH * HH + 255) / 256, 256, 0, stream>>>(Wv, WvS, HH, HH);
    k_swizzleB<<<(HH * HH + 255) / 256, 256, 0, stream>>>(We, WeS, HH, HH);
    k_swizzleB<<<(HH * HH + 255) / 256, 256, 0, stream>>>(Wo, WoS, HH, HH);
    k_swizzleB<<<(HH * FFD + 255) / 256, 256, 0, stream>>>(Wff1, Wff1S, HH, FFD);
    k_swizzleB<<<(FFD * HH + 255) / 256, 256, 0, stream>>>(Wff2, Wff2S, FFD, HH);

    // 2) node embedding gather
    k_embed<<<ewg, ewb, 0, stream>>>(x, emb, h, hb);

    // 3) radius-graph KNN
    k_knn<<<NN, 256, 0, stream>>>(pos, nbr, elen, emask);

    // 4) q/k/v projections  [8192,128] x [128,128] (WMMA bf16)
    dim3 gproj(NN / 128, HH / 16), bproj(128);
    k_gemm<0><<<gproj, bproj, 0, stream>>>(hb, WqS, nullptr, q,    nullptr, NN, HH, HH);
    k_gemm<0><<<gproj, bproj, 0, stream>>>(hb, WkS, nullptr, kprj, nullptr, NN, HH, HH);
    k_gemm<0><<<gproj, bproj, 0, stream>>>(hb, WvS, nullptr, vprj, nullptr, NN, HH, HH);

    // 5) fused RBF + e-projection (WMMA) + edge-modulated attention
    k_attn<<<NN, 128, 0, stream>>>(q, kprj, vprj, WeS, nbr, elen, emask, msg);

    // 6) t1 = h + msg @ Wo
    k_f2bf<<<ewg, ewb, 0, stream>>>(msg, msgb, EW);
    k_gemm<1><<<gproj, bproj, 0, stream>>>(msgb, WoS, h, t1, nullptr, NN, HH, HH);

    // 7) h1 = BN1(t1)
    k_bnstats<<<HH, 256, 0, stream>>>(t1, mean1, istd1);
    k_bnapply<<<ewg, ewb, 0, stream>>>(t1, mean1, istd1, bn1g, bn1b, h1, h1b);

    // 8) FFN: ffb = relu(h1 @ Wff1) (bf16 epilogue), t2 = h1 + ffb @ Wff2
    dim3 gff1(NN / 128, FFD / 16);
    k_gemm<2><<<gff1, bproj, 0, stream>>>(h1b, Wff1S, nullptr, nullptr, ffb, NN, FFD, HH);
    k_gemm<1><<<gproj, bproj, 0, stream>>>(ffb, Wff2S, h1, t2, nullptr, NN, HH, FFD);

    // 9) h2 = BN2(t2)
    k_bnstats<<<HH, 256, 0, stream>>>(t2, mean2, istd2);
    k_bnapply<<<ewg, ewb, 0, stream>>>(t2, mean2, istd2, bn2g, bn2b, h2, nullptr);

    // 10) pooling + readout head
    k_pool_readout<<<NB, HH, 0, stream>>>(h2, Wr1, br1, Wr2, br2, out);
}
